// DCWTv2Attention_5111011082700
// MI455X (gfx1250) — matrix operand order repaired and checked by reading, not compiled, not run on hardware
//
#include <hip/hip_runtime.h>
#include <cstdint>

// ---------------------------------------------------------------------------
// DCWTv2Attention for MI455X (gfx1250, wave32, WMMA + async/TDM staging).
// B=1, S=1024, E=1024, H=16, D=64, KMAX=8, WIN=32, LOG_N=11, LEAF_N=2048, NDEP=12
// ---------------------------------------------------------------------------

typedef __attribute__((ext_vector_type(16))) __bf16 v16bf;
typedef __attribute__((ext_vector_type(8)))  __bf16 v8bf;
typedef __attribute__((ext_vector_type(8)))  float  v8f;
typedef __attribute__((ext_vector_type(4)))  unsigned v4u;
typedef __attribute__((ext_vector_type(8)))  unsigned v8u;

#define S_LEN   1024
#define HEADS   16
#define DIM     64
#define EMB     1024
#define NDEP    12
#define LEAF_N  2048
#define KMAXC   8
#define PI4     0.78539816339744831f

__device__ __forceinline__ int kd_of(int d) { int v = 1 << d; return v > 8 ? 8 : v; }

__constant__ int c_offs[NDEP] = {0, 2048, 4096, 6144, 8192, 9216, 9728, 9984,
                                 10112, 10176, 10208, 10224};
#define BANK_NODES 10232

__device__ __forceinline__ float softplusf_(float x) {
  return x > 20.0f ? x : log1pf(__expf(x));
}
__device__ __forceinline__ float sigmoidf_(float x) {
  return 1.0f / (1.0f + __expf(-x));
}

// --- CDNA5 async / tensor data movement helpers (inline asm, ISA ch.10/7) ---
__device__ __forceinline__ void async_load_b128(unsigned lds_byte_addr, const void* gptr) {
  // GV mode: VDST = LDS byte address, VADDR = 64-bit global address
  asm volatile("global_load_async_to_lds_b128 %0, %1, off"
               :: "v"(lds_byte_addr), "v"(gptr) : "memory");
}
__device__ __forceinline__ void wait_asynccnt0() {
  asm volatile("s_wait_asynccnt 0x0" ::: "memory");
}
__device__ __forceinline__ void wait_tensorcnt0() {
  asm volatile("s_wait_tensorcnt 0x0" ::: "memory");
}
// Issue TDM load of a 2D bf16 tile (32 rows x 32 elems) into LDS.
//   gaddr : global byte address of tile start
//   lds   : LDS byte address of destination
//   K     : tensor row length (elements, == dim0 stride)
//   NROWS : tensor dim1 (rows)
__device__ __forceinline__ void tdm_load_tile_b16_32x32(uint64_t gaddr, unsigned lds,
                                                        unsigned K, unsigned NROWS) {
  // D# group 0 (ISA 8.3)
  v4u g0;
  g0[0] = 1u;                                   // count=1, user mode, no gather
  g0[1] = lds;                                  // lds_addr [63:32]
  g0[2] = (unsigned)(gaddr & 0xFFFFFFFFull);    // global_addr [95:64]
  g0[3] = (unsigned)((gaddr >> 32) & 0x01FFFFFFull) | (2u << 30); // addr hi | type=2
  // D# group 1 (ISA 8.4): data_size=1(2B), pad_enable, pad_interval=3(16 dw),
  // pad_amount=3(4 dw) -> LDS row stride 32+8 halves.
  v8u g1;
  g1[0] = 0x00010000u | (1u << 20) | (3u << 22) | (3u << 25);
  g1[1] = (K & 0xFFFFu) << 16;                  // tensor_dim0 lo16 in [31:16]
  g1[2] = ((K >> 16) & 0xFFFFu) | ((NROWS & 0xFFFFu) << 16); // dim0 hi | dim1 lo
  g1[3] = ((NROWS >> 16) & 0xFFFFu) | (32u << 16);           // dim1 hi | tile_dim0=32
  g1[4] = 32u;                                  // tile_dim1=32, tile_dim2=0
  g1[5] = K;                                    // tensor_dim0_stride lo32
  g1[6] = 0u;                                   // stride hi | dim1_stride lo
  g1[7] = 0u;
  asm volatile("tensor_load_to_lds %0, %1" :: "s"(g0), "s"(g1) : "memory");
}

// ---------------------------------------------------------------------------
// f32 -> bf16 conversion
// ---------------------------------------------------------------------------
__global__ void cvt_f32_bf16(const float* __restrict__ src, __bf16* __restrict__ dst, int n) {
  int i = blockIdx.x * 256 + threadIdx.x;
  if (i < n) dst[i] = (__bf16)src[i];
}

// ---------------------------------------------------------------------------
// Large GEMM with LDS staging:
//   C[M,N] = A[M,K](bf16) * W[N,K]^T(bf16) (+ bias[N])
// Block = 8 waves, 64(M) x 32(N) C tile, K-step 32, double-buffered LDS.
// A tile staged via GLOBAL_LOAD_ASYNC_TO_LDS_B128 (ASYNCcnt).
// B tile staged via Tensor Data Mover (TENSORcnt), issued by wave 0.
// ---------------------------------------------------------------------------
#define APAD 8
#define ASTR (32 + APAD)    // LDS row stride in halves (40 -> 80B, 16B aligned)
__global__ void gemm_bf16_wmma_lds(const __bf16* __restrict__ A,
                                   const __bf16* __restrict__ Bw,
                                   const float* __restrict__ bias,
                                   float* __restrict__ C,
                                   int M, int N, int K) {
  __shared__ __bf16 sA[2][64][ASTR];
  __shared__ __bf16 sB[2][32][ASTR];

  const int lane = threadIdx.x;                 // 0..31
  const int wv   = threadIdx.y;                 // 0..7
  const int tid  = wv * 32 + lane;              // 0..255
  const int m0   = blockIdx.y * 64;
  const int n0   = blockIdx.x * 32;
  const int mt   = wv >> 1;                     // 0..3
  const int nt   = wv & 1;                      // 0..1
  const int half = lane >> 4;
  const int l16  = lane & 15;

  const unsigned sA_base = (unsigned)(uintptr_t)&sA[0][0][0];
  const unsigned sB_base = (unsigned)(uintptr_t)&sB[0][0][0];

  // A-tile staging assignment: one b128 per thread (64 rows x 4 segs)
  const int arow = tid >> 2;
  const int aseg = tid & 3;
  const __bf16* agp = A + (size_t)(m0 + arow) * K + aseg * 8;

  v8f acc = {};
  int buf = 0;

  // prologue: stage k0 = 0
  async_load_b128(sA_base + (unsigned)((0 * 64 + arow) * ASTR + aseg * 8) * 2, agp);
  if (wv == 0) {
    tdm_load_tile_b16_32x32((uint64_t)(uintptr_t)(Bw + (size_t)n0 * K),
                            sB_base + 0u, (unsigned)K, (unsigned)N);
    wait_tensorcnt0();
  }
  wait_asynccnt0();
  __syncthreads();

  for (int k0 = 0; k0 < K; k0 += 32) {
    // stage next tile into the other buffer
    if (k0 + 32 < K) {
      const int nb = buf ^ 1;
      async_load_b128(sA_base + (unsigned)((nb * 64 + arow) * ASTR + aseg * 8) * 2,
                      agp + (k0 + 32));
      if (wv == 0) {
        tdm_load_tile_b16_32x32((uint64_t)(uintptr_t)(Bw + (size_t)n0 * K + (k0 + 32)),
                                sB_base + (unsigned)(nb * 32 * ASTR) * 2,
                                (unsigned)K, (unsigned)N);
      }
    }

    // fragments from LDS (ISA 7.12.2 wave32 layouts)
    const __bf16* ar = &sA[buf][mt * 16 + l16][0];
    const __bf16* br = &sB[buf][nt * 16 + l16][0];
    v8bf alo = *(const v8bf*)(ar + half * 8);
    v8bf ahi = *(const v8bf*)(ar + half * 8 + 16);
    v8bf blo = *(const v8bf*)(br + half * 8);
    v8bf bhi = *(const v8bf*)(br + half * 8 + 16);
    v16bf a, b;
#pragma unroll
    for (int i = 0; i < 8; ++i) { a[i] = alo[i]; a[i + 8] = ahi[i]; b[i] = blo[i]; b[i + 8] = bhi[i]; }
    acc = __builtin_amdgcn_wmma_f32_16x16x32_bf16(false, a, false, b,
                                                  (short)0, acc, false, false);

    if (wv == 0) wait_tensorcnt0();
    wait_asynccnt0();
    __syncthreads();
    buf ^= 1;
  }

  const int col = n0 + nt * 16 + l16;
  const float bv = bias ? bias[col] : 0.0f;
#pragma unroll
  for (int j = 0; j < 8; ++j) {
    const int row = m0 + mt * 16 + j + half * 8;
    C[(size_t)row * N + col] = acc[j] + bv;
  }
}

// ---------------------------------------------------------------------------
// Direct (register-streamed) GEMM for small K: used for qn = q @ dq_w^T + q
// ---------------------------------------------------------------------------
__global__ void gemm_bf16_wmma(const __bf16* __restrict__ A,
                               const __bf16* __restrict__ Bw,
                               const float* __restrict__ bias,
                               const float* __restrict__ addsrc,
                               float* __restrict__ C,
                               int M, int N, int K) {
  const int lane = threadIdx.x;
  const int wib  = threadIdx.y;
  const int n0   = blockIdx.x * 16;
  const int m0   = (blockIdx.y * 4 + wib) * 16;
  if (m0 >= M || n0 >= N) return;

  const int half = lane >> 4;
  const int l16  = lane & 15;

  const __bf16* aptr = A  + (size_t)(m0 + l16) * K + half * 16;
  const __bf16* bptr = Bw + (size_t)(n0 + l16) * K + half * 16;

  v8f acc = {};
  for (int k0 = 0; k0 < K; k0 += 32) {
    __builtin_prefetch(aptr + k0 + 128, 0, 1);
    __builtin_prefetch(bptr + k0 + 128, 0, 1);
    v8bf alo = *(const v8bf*)(aptr + k0);
    v8bf ahi = *(const v8bf*)(aptr + k0 + 8);
    v16bf a;
#pragma unroll
    for (int i = 0; i < 8; ++i) { a[i] = alo[i]; a[i + 8] = ahi[i]; }
    v16bf b = *(const v16bf*)(bptr + k0);
    acc = __builtin_amdgcn_wmma_f32_16x16x32_bf16(false, a, false, b,
                                                  (short)0, acc, false, false);
  }

  const int col = n0 + l16;
  const float bv = bias ? bias[col] : 0.0f;
#pragma unroll
  for (int j = 0; j < 8; ++j) {
    const int row = m0 + j + half * 8;
    float o = acc[j] + bv;
    if (addsrc) o += addsrc[(size_t)row * N + col];
    C[(size_t)row * N + col] = o;
  }
}

// ---------------------------------------------------------------------------
// bank level 0: first S nodes = v, rest zero.  bank[node][h][dd]
// ---------------------------------------------------------------------------
__global__ void bank_init(const float* __restrict__ v, float* __restrict__ bank) {
  int i = blockIdx.x * 256 + threadIdx.x;
  int node = i >> 10;
  int r    = i & 1023;
  bank[i] = (node < S_LEN) ? v[(size_t)node * 1024 + r] : 0.0f;
}

// ---------------------------------------------------------------------------
// One segment-tree merge level. grid = (2048>>d, H), block = 64 (t = channel).
// ---------------------------------------------------------------------------
__global__ void merge_level(float* __restrict__ bank,
                            const float* __restrict__ gate_lw, const float* __restrict__ gate_lb,
                            const float* __restrict__ gate_rw, const float* __restrict__ gate_rb,
                            const float* __restrict__ parent_q,
                            const float* __restrict__ ln_w, const float* __restrict__ ln_b,
                            const float* __restrict__ skip_a, const float* __restrict__ skip_w,
                            const float* __restrict__ wfreq, const float* __restrict__ wdamp,
                            const float* __restrict__ wphase,
                            int d) {
  const int lp = blockIdx.x;
  const int h  = blockIdx.y;
  const int dd = threadIdx.x;

  const int kprev  = kd_of(d - 1);
  const int kp     = kd_of(d);
  const int nslots = 2 * kprev;
  const int offc   = c_offs[d - 1];
  const int offp   = c_offs[d];

  __shared__ float s_gin[128];
  __shared__ float s_b[16][64];
  __shared__ float s_att[8][16];
  __shared__ float s_p[8][64];
  __shared__ float s_mu[8], s_rs[8];

  const float ang   = wfreq[h] + wphase[h] + (float)d * PI4;
  const float decay = __expf(-softplusf_(wdamp[h]));
  const float pr  = decay * __cosf(ang);
  const float pii = decay * __sinf(ang);
  const int partner = dd ^ 32;

  float fl[KMAXC], rr[KMAXC];
  float lm = 0.0f, rm = 0.0f;
  for (int k = 0; k < kprev; ++k) {
    const float* pl = bank + ((size_t)(offc + (2 * lp)     * kprev + k) * HEADS + h) * DIM;
    const float* prt= bank + ((size_t)(offc + (2 * lp + 1) * kprev + k) * HEADS + h) * DIM;
    fl[k] = pl[dd];
    const float frv = prt[dd];
    const float frp = prt[partner];
    rr[k] = (dd < 32) ? (pr * frv - pii * frp) : (pii * frp + pr * frv);
    lm += fl[k]; rm += rr[k];
  }
  lm /= (float)kprev; rm /= (float)kprev;
  s_gin[dd] = lm; s_gin[64 + dd] = rm;
  __syncthreads();

  const float* glw = gate_lw + ((size_t)d * 64 + dd) * 128;
  const float* grw = gate_rw + ((size_t)d * 64 + dd) * 128;
  float accl = gate_lb[d * 64 + dd];
  float accr = gate_rb[d * 64 + dd];
  for (int j = 0; j < 128; ++j) { accl += glw[j] * s_gin[j]; accr += grw[j] * s_gin[j]; }
  const float gl = sigmoidf_(accl);
  const float gr = sigmoidf_(accr);
  for (int k = 0; k < kprev; ++k) {
    s_b[k][dd]         = fl[k] * gl;
    s_b[kprev + k][dd] = rr[k] * gr;
  }
  __syncthreads();

  for (int e = dd; e < kp * nslots; e += 64) {
    const int qi = e / nslots, kk = e % nslots;
    const float* pq = parent_q + ((size_t)d * KMAXC + qi) * DIM;
    float sacc = 0.0f;
    for (int j = 0; j < 64; ++j) sacc += pq[j] * s_b[kk][j];
    s_att[qi][kk] = sacc * 0.125f;
  }
  __syncthreads();
  if (dd < kp) {
    float mx = -1e30f;
    for (int kk = 0; kk < nslots; ++kk) mx = fmaxf(mx, s_att[dd][kk]);
    float sum = 0.0f;
    for (int kk = 0; kk < nslots; ++kk) { float e = __expf(s_att[dd][kk] - mx); s_att[dd][kk] = e; sum += e; }
    const float inv = 1.0f / sum;
    for (int kk = 0; kk < nslots; ++kk) s_att[dd][kk] *= inv;
  }
  __syncthreads();

  for (int qi = 0; qi < kp; ++qi) {
    float sacc = 0.0f;
    for (int kk = 0; kk < nslots; ++kk) sacc += s_att[qi][kk] * s_b[kk][dd];
    s_p[qi][dd] = sacc;
  }
  __syncthreads();
  if (dd < kp) {
    float mu = 0.0f;
    for (int j = 0; j < 64; ++j) mu += s_p[dd][j];
    mu *= (1.0f / 64.0f);
    float var = 0.0f;
    for (int j = 0; j < 64; ++j) { float t = s_p[dd][j] - mu; var += t * t; }
    var *= (1.0f / 64.0f);
    s_mu[dd] = mu; s_rs[dd] = rsqrtf(var + 1e-5f);
  }
  __syncthreads();

  const float* sw = skip_w + ((size_t)d * 64 + dd) * 64;
  float sk = 0.0f;
  for (int j = 0; j < 64; ++j) sk += sw[j] * s_gin[j];
  const float siga = sigmoidf_(skip_a[d]);
  const float lw = ln_w[d * 64 + dd], lb = ln_b[d * 64 + dd];
  for (int qi = 0; qi < kp; ++qi) {
    const float o = (s_p[qi][dd] - s_mu[qi]) * s_rs[qi] * lw + lb + siga * sk;
    bank[((size_t)(offp + lp * kp + qi) * HEADS + h) * DIM + dd] = o;
  }
}

// ---------------------------------------------------------------------------
// Tree attention over segment cover. grid = (S, H), block = 64.
// ---------------------------------------------------------------------------
#define MAXM 192
__device__ __forceinline__ int push_node(int node, int cnt, int* idx, int* dep) {
  const int lvl = 31 - __builtin_clz(node);
  const int d   = 11 - lvl;
  const int kdd = kd_of(d);
  const int lp  = node - (LEAF_N >> d);
  for (int k = 0; k < kdd; ++k) { idx[cnt] = c_offs[d] + lp * kdd + k; dep[cnt] = d; cnt++; }
  return cnt;
}

__global__ void tree_attn(const float* __restrict__ qn,
                          const float* __restrict__ bank,
                          const float* __restrict__ dq_temp,
                          float* __restrict__ tree) {
  const int s = blockIdx.x, h = blockIdx.y, dd = threadIdx.x;

  __shared__ int   s_idx[MAXM];
  __shared__ int   s_dep[MAXM];
  __shared__ float s_sc[MAXM];
  __shared__ float s_scale[NDEP];
  __shared__ int   s_cnt;

  if (dd < NDEP) s_scale[dd] = 1.0f / ((softplusf_(dq_temp[dd]) + 1e-6f) * 8.0f);
  if (dd == 0) {
    int cnt = 0;
    int l = LEAF_N, r = LEAF_N + s;
    while (l < r) {
      if (l & 1) { cnt = push_node(l, cnt, s_idx, s_dep); l++; }
      if (r & 1) { r--; cnt = push_node(r, cnt, s_idx, s_dep); }
      l >>= 1; r >>= 1;
    }
    s_cnt = cnt;
  }
  __syncthreads();

  const int cnt = s_cnt;
  for (int m = dd; m < cnt; m += 64) {
    const int d = s_dep[m];
    const float* qv = qn + ((size_t)d * S_LEN * HEADS + (size_t)s * HEADS + h) * DIM;
    const float* bv = bank + ((size_t)s_idx[m] * HEADS + h) * DIM;
    float acc = 0.0f;
    for (int j = 0; j < 64; ++j) acc += qv[j] * bv[j];
    s_sc[m] = acc * s_scale[d];
  }
  __syncthreads();
  if (dd == 0 && cnt > 0) {
    float mx = -1e30f;
    for (int m = 0; m < cnt; ++m) mx = fmaxf(mx, s_sc[m]);
    float sum = 0.0f;
    for (int m = 0; m < cnt; ++m) { float e = __expf(s_sc[m] - mx); s_sc[m] = e; sum += e; }
    const float inv = 1.0f / sum;
    for (int m = 0; m < cnt; ++m) s_sc[m] *= inv;
  }
  __syncthreads();

  float o = 0.0f;
  for (int m = 0; m < cnt; ++m)
    o += s_sc[m] * bank[((size_t)s_idx[m] * HEADS + h) * DIM + dd];
  tree[((size_t)s * HEADS + h) * DIM + dd] = (s > 0) ? o : 0.0f;
}

// ---------------------------------------------------------------------------
// Local causal windowed attention (WIN=32). grid = (S, H), block = 64.
// ---------------------------------------------------------------------------
__global__ void local_attn(const float* __restrict__ q,
                           const float* __restrict__ kl,
                           const float* __restrict__ vl,
                           float* __restrict__ outl) {
  const int s = blockIdx.x, h = blockIdx.y, dd = threadIdx.x;
  __shared__ float s_sc[32];
  const int jstart = (s >= 31) ? s - 31 : 0;
  const int nj = s - jstart + 1;
  if (dd < nj) {
    const int j = jstart + dd;
    const float* qv = q  + ((size_t)s * HEADS + h) * DIM;
    const float* kv = kl + ((size_t)j * HEADS + h) * DIM;
    float a = 0.0f;
    for (int i = 0; i < 64; ++i) a += qv[i] * kv[i];
    s_sc[dd] = a * 0.125f;
  }
  __syncthreads();
  if (dd == 0) {
    float mx = -1e30f;
    for (int t = 0; t < nj; ++t) mx = fmaxf(mx, s_sc[t]);
    float sum = 0.0f;
    for (int t = 0; t < nj; ++t) { float e = __expf(s_sc[t] - mx); s_sc[t] = e; sum += e; }
    const float inv = 1.0f / sum;
    for (int t = 0; t < nj; ++t) s_sc[t] *= inv;
  }
  __syncthreads();
  float o = 0.0f;
  for (int t = 0; t < nj; ++t)
    o += s_sc[t] * vl[((size_t)(jstart + t) * HEADS + h) * DIM + dd];
  outl[((size_t)s * HEADS + h) * DIM + dd] = o;
}

// ---------------------------------------------------------------------------
// combine: sigmoid(alpha)*local + (1-..)*tree -> bf16 for final GEMM
// ---------------------------------------------------------------------------
__global__ void combine_bf16(const float* __restrict__ localb,
                             const float* __restrict__ treeb,
                             const float* __restrict__ alpha,
                             __bf16* __restrict__ cb, int n) {
  int i = blockIdx.x * 256 + threadIdx.x;
  if (i >= n) return;
  const float a = 1.0f / (1.0f + __expf(-alpha[0]));
  cb[i] = (__bf16)(a * localb[i] + (1.0f - a) * treeb[i]);
}

// ---------------------------------------------------------------------------
// host launcher
// ---------------------------------------------------------------------------
extern "C" void kernel_launch(void* const* d_in, const int* in_sizes, int n_in,
                              void* d_out, int out_size, void* d_ws, size_t ws_size,
                              hipStream_t stream) {
  const float* x      = (const float*)d_in[0];
  const float* wq     = (const float*)d_in[1];
  const float* bq     = (const float*)d_in[2];
  const float* wv     = (const float*)d_in[3];
  const float* bv     = (const float*)d_in[4];
  const float* wkl    = (const float*)d_in[5];
  const float* bkl    = (const float*)d_in[6];
  const float* wvl    = (const float*)d_in[7];
  const float* bvl    = (const float*)d_in[8];
  const float* wo     = (const float*)d_in[9];
  const float* bo     = (const float*)d_in[10];
  const float* dq_w   = (const float*)d_in[11];
  const float* dq_temp= (const float*)d_in[12];
  const float* wfreq  = (const float*)d_in[13];
  const float* wdamp  = (const float*)d_in[14];
  const float* wphase = (const float*)d_in[15];
  const float* g_lw   = (const float*)d_in[16];
  const float* g_lb   = (const float*)d_in[17];
  const float* g_rw   = (const float*)d_in[18];
  const float* g_rb   = (const float*)d_in[19];
  const float* pq     = (const float*)d_in[20];
  const float* ln_w   = (const float*)d_in[21];
  const float* ln_b   = (const float*)d_in[22];
  const float* skip_a = (const float*)d_in[23];
  const float* skip_w = (const float*)d_in[24];
  const float* halpha = (const float*)d_in[25];

  size_t off = 0;
  auto alloc = [&](size_t bytes) -> char* {
    char* p = (char*)d_ws + off;
    off += (bytes + 255) & ~(size_t)255;
    return p;
  };
  const size_t EE = (size_t)EMB * EMB;
  __bf16* xb   = (__bf16*)alloc(EE * 2);
  __bf16* wqb  = (__bf16*)alloc(EE * 2);
  __bf16* wvb  = (__bf16*)alloc(EE * 2);
  __bf16* wklb = (__bf16*)alloc(EE * 2);
  __bf16* wvlb = (__bf16*)alloc(EE * 2);
  __bf16* wob  = (__bf16*)alloc(EE * 2);
  __bf16* dqwb = (__bf16*)alloc((size_t)NDEP * 64 * 64 * 2);
  float*  qf   = (float*)alloc(EE * 4);
  float*  vf   = (float*)alloc(EE * 4);
  float*  klf  = (float*)alloc(EE * 4);
  float*  vlf  = (float*)alloc(EE * 4);
  __bf16* qb   = (__bf16*)alloc(EE * 2);
  float*  qn   = (float*)alloc((size_t)NDEP * EE * 4);
  float*  bank = (float*)alloc((size_t)BANK_NODES * HEADS * DIM * 4);
  float*  treeb= (float*)alloc(EE * 4);
  float*  locb = (float*)alloc(EE * 4);
  __bf16* cb   = (__bf16*)alloc(EE * 2);

  const dim3 blk256(256);
  hipLaunchKernelGGL(cvt_f32_bf16, dim3((EE + 255) / 256), blk256, 0, stream, x,   xb,   (int)EE);
  hipLaunchKernelGGL(cvt_f32_bf16, dim3((EE + 255) / 256), blk256, 0, stream, wq,  wqb,  (int)EE);
  hipLaunchKernelGGL(cvt_f32_bf16, dim3((EE + 255) / 256), blk256, 0, stream, wv,  wvb,  (int)EE);
  hipLaunchKernelGGL(cvt_f32_bf16, dim3((EE + 255) / 256), blk256, 0, stream, wkl, wklb, (int)EE);
  hipLaunchKernelGGL(cvt_f32_bf16, dim3((EE + 255) / 256), blk256, 0, stream, wvl, wvlb, (int)EE);
  hipLaunchKernelGGL(cvt_f32_bf16, dim3((EE + 255) / 256), blk256, 0, stream, wo,  wob,  (int)EE);
  hipLaunchKernelGGL(cvt_f32_bf16, dim3((NDEP * 64 * 64 + 255) / 256), blk256, 0, stream,
                     dq_w, dqwb, NDEP * 64 * 64);

  // big projections: LDS-staged WMMA (async + TDM)
  const dim3 lblk(32, 8);
  const dim3 lgrd(EMB / 32, S_LEN / 64);
  hipLaunchKernelGGL(gemm_bf16_wmma_lds, lgrd, lblk, 0, stream, xb, wqb,  bq,  qf,  S_LEN, EMB, EMB);
  hipLaunchKernelGGL(gemm_bf16_wmma_lds, lgrd, lblk, 0, stream, xb, wvb,  bv,  vf,  S_LEN, EMB, EMB);
  hipLaunchKernelGGL(gemm_bf16_wmma_lds, lgrd, lblk, 0, stream, xb, wklb, bkl, klf, S_LEN, EMB, EMB);
  hipLaunchKernelGGL(gemm_bf16_wmma_lds, lgrd, lblk, 0, stream, xb, wvlb, bvl, vlf, S_LEN, EMB, EMB);

  // qn[d] = q + q @ dq_w[d]^T  (direct WMMA, K=64)
  hipLaunchKernelGGL(cvt_f32_bf16, dim3((EE + 255) / 256), blk256, 0, stream, qf, qb, (int)EE);
  const dim3 gblk(32, 4);
  const dim3 qgrd(64 / 16, (S_LEN * HEADS) / 64);
  for (int d = 0; d < NDEP; ++d) {
    hipLaunchKernelGGL(gemm_bf16_wmma, qgrd, gblk, 0, stream,
                       qb, dqwb + (size_t)d * 64 * 64, (const float*)nullptr, qf,
                       qn + (size_t)d * EE, S_LEN * HEADS, 64, 64);
  }

  // segment-tree bank
  hipLaunchKernelGGL(bank_init, dim3((LEAF_N * HEADS * DIM) / 256), blk256, 0, stream, vf, bank);
  for (int d = 1; d < NDEP; ++d) {
    hipLaunchKernelGGL(merge_level, dim3(LEAF_N >> d, HEADS), dim3(64), 0, stream,
                       bank, g_lw, g_lb, g_rw, g_rb, pq, ln_w, ln_b, skip_a, skip_w,
                       wfreq, wdamp, wphase, d);
  }

  hipLaunchKernelGGL(tree_attn,  dim3(S_LEN, HEADS), dim3(64), 0, stream, qn, bank, dq_temp, treeb);
  hipLaunchKernelGGL(local_attn, dim3(S_LEN, HEADS), dim3(64), 0, stream, qf, klf, vlf, locb);

  hipLaunchKernelGGL(combine_bf16, dim3((EE + 255) / 256), blk256, 0, stream,
                     locb, treeb, halpha, cb, (int)EE);
  hipLaunchKernelGGL(gemm_bf16_wmma_lds, lgrd, lblk, 0, stream,
                     cb, wob, bo, (float*)d_out, S_LEN, EMB, EMB);
}
